// ShiftDecoderNet_32512902431267
// MI455X (gfx1250) — matrix operand order repaired and checked by reading, not compiled, not run on hardware
//
#include <hip/hip_runtime.h>

// ---------------------------------------------------------------------------
// Fused ShiftDecoderNet for gfx1250 (MI455X), TDM edition:
//   prep kernel: W1/W2/W3 f32 -> f16 into d_ws (one-time per call, 192 KB)
//   main kernel: shift_bits -> [GEMM1 WMMA] -> LN+ReLU -> [GEMM2] -> LN+ReLU
//                -> [GEMM3] -> softmax(64) -> causal Toeplitz -> out (f32)
// Weight k-tiles are streamed into double-buffered LDS by the Tensor Data
// Mover (tensor_load_to_lds, TENSORcnt), with TDM padding producing the
// bank-conflict-free LDS pitch. One block = 256 threads = 8 wave32 owns 128
// batch rows; wave w owns rows 16w..16w+15 (no cross-wave activation hazards).
// ---------------------------------------------------------------------------

typedef __attribute__((ext_vector_type(16))) _Float16 v16h;
typedef __attribute__((ext_vector_type(8)))  _Float16 v8h;
typedef __attribute__((ext_vector_type(8)))  float    v8f;
typedef __attribute__((ext_vector_type(4)))  float    v4f;
typedef __attribute__((ext_vector_type(4)))  unsigned int v4u;
typedef __attribute__((ext_vector_type(8)))  unsigned int v8u;

#define BITS        64
#define HID         256
#define BM          128         // batch rows per block
#define NTHREADS    256         // 8 wave32
#define ACT_STRIDE  264         // halves per activation row (256 + 8 pad)
#define W_STRIDE    40          // halves per weight-tile row (32 + 8 pad, from TDM pad)
#define P_STRIDE    68          // floats per prob row
#define APAD_STRIDE 132         // floats per padded a row: 64 zeros + 64 data + 4
#define LN_EPS      1e-5f

// LDS layout (disjoint regions, 206 KB < 320 KB WGP LDS)
#define ACT_OFF    0
#define ACT_BYTES  (BM * ACT_STRIDE * 2)        // 67584
#define WT_BYTES   (HID * W_STRIDE * 2)         // 20480
#define WT0_OFF    ACT_BYTES                    // 67584
#define WT1_OFF    (WT0_OFF + WT_BYTES)         // 88064
#define P_OFF      (WT1_OFF + WT_BYTES)         // 108544
#define P_BYTES    (BM * P_STRIDE * 4)          // 34816
#define APAD_OFF   (P_OFF + P_BYTES)            // 143360
#define APAD_BYTES (BM * APAD_STRIDE * 4)       // 67584
#define SMEM_BYTES (APAD_OFF + APAD_BYTES)      // 210944

// d_ws layout in halves: W1h[256*64] | W2h[256*256] | W3h[64*256] (196608 B)
#define W1H_OFF 0
#define W2H_OFF 16384
#define W3H_OFF 81920
#define WSH_ELEMS 98304

__device__ __forceinline__ float redsum16(float v) {
    v += __shfl_xor(v, 1, 32);
    v += __shfl_xor(v, 2, 32);
    v += __shfl_xor(v, 4, 32);
    v += __shfl_xor(v, 8, 32);
    return v;
}
__device__ __forceinline__ float redmax16(float v) {
    v = fmaxf(v, __shfl_xor(v, 1, 32));
    v = fmaxf(v, __shfl_xor(v, 2, 32));
    v = fmaxf(v, __shfl_xor(v, 4, 32));
    v = fmaxf(v, __shfl_xor(v, 8, 32));
    return v;
}

// --------------------------------------------------------------------------
// Tensor Data Mover: 2D tile copy global -> LDS with optional LDS padding.
// D# built per CDNA5 ISA 8.3/8.4 (group0 128b + group1 256b; groups 2/3
// NULL => 2-operand tensor_load_to_lds form, tile up to 2D). data_size=1B.
// Tracked by TENSORcnt; EXEC-independent; issue from one wave only.
// --------------------------------------------------------------------------
__device__ __forceinline__ void tdm_load_2d(unsigned lds_addr, const void* gptr,
                                            unsigned tile_x_bytes, unsigned rows,
                                            unsigned row_stride_bytes,
                                            unsigned pad_int_code, unsigned pad_amt_code) {
    unsigned long long ga = (unsigned long long)gptr;
    v4u g0;
    g0[0] = 1u;                                            // count=1, no gather
    g0[1] = lds_addr;                                      // lds_addr [63:32]
    g0[2] = (unsigned)ga;                                  // global_addr[31:0]
    g0[3] = (unsigned)((ga >> 32) & 0x01ffffffu)           // global_addr[56:32]
          | (2u << 30);                                    // type=2 (image)
    v8u g1;
    g1[0] = (1u << 20)                                     // pad_enable
          | (pad_int_code << 22)                           // pad_interval
          | (pad_amt_code << 25);                          // pad_amount
    g1[1] = (tile_x_bytes & 0xffffu) << 16;                // tensor_dim0 lo16
    g1[2] = ((tile_x_bytes >> 16) & 0xffffu)               // tensor_dim0 hi16
          | ((rows & 0xffffu) << 16);                      // tensor_dim1 lo16
    g1[3] = ((rows >> 16) & 0xffffu)                       // tensor_dim1 hi16
          | ((tile_x_bytes & 0xffffu) << 16);              // tile_dim0
    g1[4] = rows & 0xffffu;                                // tile_dim1 (tile_dim2=0)
    g1[5] = row_stride_bytes;                              // tensor_dim0_stride lo32
    g1[6] = 0u;                                            // stride0 hi16 | stride1 lo16
    g1[7] = 0u;                                            // stride1 hi32
    asm volatile("tensor_load_to_lds %0, %1" :: "s"(g0), "s"(g1) : "memory");
}

// --------------------------------------------------------------------------
// Tiled GEMM stage: act[BM x KDIM] (f16 LDS) @ Wh^T, Wh = f16 [NT*16 x KDIM]
// row-major in global (pre-converted). Weight k-tiles (NT*16 x 32 halves)
// are TDM-streamed into double-buffered LDS; pipeline:
//   wait(tile k) -> barrier -> issue(tile k+1, other buf) -> WMMA(tile k)
// --------------------------------------------------------------------------
template <int KDIM, int NT>
__device__ __forceinline__ void gemm_stage(const _Float16* __restrict__ Wh,
                                           const _Float16* act,
                                           const _Float16* wt0, const _Float16* wt1,
                                           unsigned wt0_lds, unsigned wt1_lds,
                                           v8f* acc, int lo, int hi, int wv_s) {
    constexpr int NSTEPS = KDIM / 32;
#pragma unroll
    for (int t = 0; t < NT; ++t)
#pragma unroll
        for (int j = 0; j < 8; ++j) acc[t][j] = 0.0f;

    // preload tile 0 into wt0 (pad codes: 16 DW interval, 4 DW pad -> 80B pitch)
    if (wv_s == 0)
        tdm_load_2d(wt0_lds, Wh, 64u, (unsigned)(NT * 16), (unsigned)(KDIM * 2), 3u, 3u);

    for (int ks = 0; ks < NSTEPS; ++ks) {
        if (wv_s == 0) __builtin_amdgcn_s_wait_tensorcnt(0);
        __syncthreads();   // tile ks visible; everyone done with buf[ks^1] (prev iter)
        if (wv_s == 0 && ks + 1 < NSTEPS)
            tdm_load_2d((ks & 1) ? wt0_lds : wt1_lds,
                        (const char*)Wh + (size_t)(ks + 1) * 64,
                        64u, (unsigned)(NT * 16), (unsigned)(KDIM * 2), 3u, 3u);

        const _Float16* wt = (ks & 1) ? wt1 : wt0;
        // A fragment (16x32 f16): lane L -> row m=16w+L%16,
        // halves [ks*32 + hi*8 .. +7] and [ks*32 + 16 + hi*8 .. +7]
        const _Float16* ap = act + (size_t)((wv_s * 16 + lo)) * ACT_STRIDE + ks * 32 + hi * 8;
        v8h a0 = *(const v8h*)ap;
        v8h a1 = *(const v8h*)(ap + 16);
        v16h a = __builtin_shufflevector(a0, a1, 0, 1, 2, 3, 4, 5, 6, 7,
                                         8, 9, 10, 11, 12, 13, 14, 15);
#pragma unroll
        for (int nt = 0; nt < NT; ++nt) {
            const _Float16* bp = wt + (size_t)(nt * 16 + lo) * W_STRIDE + hi * 8;
            v8h b0 = *(const v8h*)bp;
            v8h b1 = *(const v8h*)(bp + 16);
            v16h b = __builtin_shufflevector(b0, b1, 0, 1, 2, 3, 4, 5, 6, 7,
                                             8, 9, 10, 11, 12, 13, 14, 15);
            acc[nt] = __builtin_amdgcn_wmma_f32_16x16x32_f16(
                false, a, false, b, (short)0, acc[nt], false, false);
        }
    }
}

// bias + LayerNorm(256) + ReLU epilogue, storing f16 to the wave's own rows.
__device__ __forceinline__ void ln_relu_store(v8f* acc, _Float16* act,
                                              const float* __restrict__ bias,
                                              const float* __restrict__ gamma,
                                              const float* __restrict__ beta,
                                              int lo, int hi, int wv_s) {
    float bv[16], gv[16], ev[16];
#pragma unroll
    for (int nt = 0; nt < 16; ++nt) {
        int n = nt * 16 + lo;
        bv[nt] = bias[n]; gv[nt] = gamma[n]; ev[nt] = beta[n];
    }
#pragma unroll
    for (int r = 0; r < 8; ++r) {
        float s1 = 0.f, s2 = 0.f;
#pragma unroll
        for (int nt = 0; nt < 16; ++nt) {
            float x = acc[nt][r] + bv[nt];
            s1 += x; s2 += x * x;
        }
        s1 = redsum16(s1);
        s2 = redsum16(s2);
        float mu   = s1 * (1.0f / HID);
        float var  = s2 * (1.0f / HID) - mu * mu;
        float rstd = rsqrtf(var + LN_EPS);
        int   m    = wv_s * 16 + hi * 8 + r;
#pragma unroll
        for (int nt = 0; nt < 16; ++nt) {
            float x = acc[nt][r] + bv[nt];
            float y = (x - mu) * rstd * gv[nt] + ev[nt];
            act[(size_t)m * ACT_STRIDE + nt * 16 + lo] = (_Float16)fmaxf(y, 0.0f);
        }
    }
}

__global__ void prep_weights_f16(const float* __restrict__ W1,
                                 const float* __restrict__ W2,
                                 const float* __restrict__ W3,
                                 _Float16* __restrict__ wsh) {
    int e = blockIdx.x * 256 + threadIdx.x;
    if (e < W2H_OFF)            wsh[e] = (_Float16)W1[e];
    else if (e < W3H_OFF)       wsh[e] = (_Float16)W2[e - W2H_OFF];
    else if (e < WSH_ELEMS)     wsh[e] = (_Float16)W3[e - W3H_OFF];
}

__global__ void __launch_bounds__(NTHREADS)
shiftdecoder_fused_kernel(const float* __restrict__ a_bits,
                          const float* __restrict__ shift_bits,
                          const _Float16* __restrict__ wsh,
                          const float* __restrict__ b1, const float* __restrict__ g1v,
                          const float* __restrict__ be1,
                          const float* __restrict__ b2, const float* __restrict__ g2v,
                          const float* __restrict__ be2,
                          const float* __restrict__ b3,
                          float* __restrict__ out) {
    extern __shared__ unsigned char smem[];
    _Float16* act  = (_Float16*)(smem + ACT_OFF);
    _Float16* wt0  = (_Float16*)(smem + WT0_OFF);
    _Float16* wt1  = (_Float16*)(smem + WT1_OFF);
    float*    Pp   = (float*)(smem + P_OFF);
    float*    Apad = (float*)(smem + APAD_OFF);

    const int    tid  = threadIdx.x;
    const int    lane = tid & 31;
    const int    lo   = lane & 15;
    const int    hi   = lane >> 4;
    const int    wv_s = __builtin_amdgcn_readfirstlane(tid >> 5);  // uniform wave id
    const size_t row0 = (size_t)blockIdx.x * BM;

    // LDS region offsets for TDM (flat LDS ptr low 32 bits = wave-LDS offset)
    const unsigned lds0     = (unsigned)(unsigned long long)(void*)smem;
    const unsigned wt0_lds  = lds0 + WT0_OFF;
    const unsigned wt1_lds  = lds0 + WT1_OFF;
    const unsigned apad_lds = lds0 + APAD_OFF + 64 * 4;   // data starts after 64 zeros

    // warm L2 for a_bits (consumed only in the final phase)
    __builtin_prefetch(a_bits + (row0 + (size_t)(tid >> 1)) * BITS + (size_t)(tid & 1) * 32, 0, 0);

    // ---- shift_bits tile (f32 -> f16) into activation buffer cols 0..63
    for (int e = tid; e < BM * BITS; e += NTHREADS) {
        int r = e >> 6, c = e & 63;
        act[(size_t)r * ACT_STRIDE + c] = (_Float16)shift_bits[(row0 + r) * BITS + c];
    }

    v8f acc[16];

    // ---- stage 1: h1 = relu(LN(shift @ W1^T + b1))
    gemm_stage<BITS, 16>(wsh + W1H_OFF, act, wt0, wt1, wt0_lds, wt1_lds, acc, lo, hi, wv_s);
    ln_relu_store(acc, act, b1, g1v, be1, lo, hi, wv_s);

    // ---- stage 2: h2 = relu(LN(h1 @ W2^T + b2))
    gemm_stage<HID, 16>(wsh + W2H_OFF, act, wt0, wt1, wt0_lds, wt1_lds, acc, lo, hi, wv_s);
    ln_relu_store(acc, act, b2, g2v, be2, lo, hi, wv_s);

    // ---- stage 3: logits = h2 @ W3^T + b3   (64 outputs -> 4 tiles)
    gemm_stage<HID, 4>(wsh + W3H_OFF, act, wt0, wt1, wt0_lds, wt1_lds, acc, lo, hi, wv_s);

    // ---- TDM-fetch a_bits rows into Apad[.][64..128) (pad: 64 DW data, 68 DW skip)
    if (wv_s == 0)
        tdm_load_2d(apad_lds, a_bits + row0 * BITS, 256u, BM, 256u, 5u, 67u);

    // zero prefix Apad[.][0..64) (causal zero-padding for the Toeplitz phase)
    for (int e = tid; e < BM * 64; e += NTHREADS) {
        int r = e >> 6, c = e & 63;
        Apad[(size_t)r * APAD_STRIDE + c] = 0.0f;
    }

    // ---- softmax over the 64 logits per row, write probs to LDS
    {
        float bv[4];
#pragma unroll
        for (int nt = 0; nt < 4; ++nt) bv[nt] = b3[nt * 16 + lo];
#pragma unroll
        for (int r = 0; r < 8; ++r) {
            int   m = wv_s * 16 + hi * 8 + r;
            float l[4];
            float mx = -3.0e38f;
#pragma unroll
            for (int nt = 0; nt < 4; ++nt) {
                l[nt] = acc[nt][r] + bv[nt];
                mx = fmaxf(mx, l[nt]);
            }
            mx = redmax16(mx);
            float s = 0.f;
#pragma unroll
            for (int nt = 0; nt < 4; ++nt) {
                l[nt] = __expf(l[nt] - mx);
                s += l[nt];
            }
            s = redsum16(s);
            float inv = 1.0f / s;
#pragma unroll
            for (int nt = 0; nt < 4; ++nt)
                Pp[(size_t)m * P_STRIDE + nt * 16 + lo] = l[nt] * inv;
        }
    }

    if (wv_s == 0) __builtin_amdgcn_s_wait_tensorcnt(0);
    __syncthreads();   // probs + padded a rows visible

    // ---- causal Toeplitz via register tiles:
    // out[i] = sum_{s=0..63} P[s] * Apad[i - s + 64]   (Apad zero-padded)
    // 8-output x 4-tap tiles: 16 LDS dwords per 32 FMAs.
    {
        int row = tid >> 1;
        int ib  = (tid & 1) * 32;
        const float* Pr = Pp + (size_t)row * P_STRIDE;
        const float* Ar = Apad + (size_t)row * APAD_STRIDE;
        float* orow = out + (row0 + row) * BITS;
#pragma unroll
        for (int it = 0; it < 4; ++it) {
            int ibase = ib + it * 8;
            float accT[8];
#pragma unroll
            for (int ii = 0; ii < 8; ++ii) accT[ii] = 0.f;
            for (int st = 0; st < 16; ++st) {
                v4f p4 = *(const v4f*)(Pr + st * 4);
                const float* wb = Ar + (ibase + 60 - st * 4);   // 16B-aligned
                v4f w0 = *(const v4f*)wb;
                v4f w1 = *(const v4f*)(wb + 4);
                v4f w2 = *(const v4f*)(wb + 8);
                float w[12] = { w0[0], w0[1], w0[2], w0[3],
                                w1[0], w1[1], w1[2], w1[3],
                                w2[0], w2[1], w2[2], w2[3] };
#pragma unroll
                for (int k = 0; k < 4; ++k)
#pragma unroll
                    for (int ii = 0; ii < 8; ++ii)
                        accT[ii] = fmaf(p4[k], w[ii + 4 - k], accT[ii]);
            }
            v4f o0 = { accT[0], accT[1], accT[2], accT[3] };
            v4f o1 = { accT[4], accT[5], accT[6], accT[7] };
            *(v4f*)(orow + ibase)     = o0;
            *(v4f*)(orow + ibase + 4) = o1;
        }
    }
}

extern "C" void kernel_launch(void* const* d_in, const int* in_sizes, int n_in,
                              void* d_out, int out_size, void* d_ws, size_t ws_size,
                              hipStream_t stream) {
    const float* a_bits     = (const float*)d_in[0];
    const float* shift_bits = (const float*)d_in[1];
    const float* W1  = (const float*)d_in[2];
    const float* b1  = (const float*)d_in[3];
    const float* g1v = (const float*)d_in[4];
    const float* be1 = (const float*)d_in[5];
    const float* W2  = (const float*)d_in[6];
    const float* b2  = (const float*)d_in[7];
    const float* g2v = (const float*)d_in[8];
    const float* be2 = (const float*)d_in[9];
    const float* W3  = (const float*)d_in[10];
    const float* b3  = (const float*)d_in[11];
    float* out = (float*)d_out;

    _Float16* wsh = (_Float16*)d_ws;    // needs 196608 B of d_ws

    const int B    = in_sizes[0] / BITS;   // 32768
    const int grid = B / BM;               // 256 blocks

    prep_weights_f16<<<(WSH_ELEMS + 255) / 256, 256, 0, stream>>>(W1, W2, W3, wsh);
    shiftdecoder_fused_kernel<<<grid, NTHREADS, SMEM_BYTES, stream>>>(
        a_bits, shift_bits, wsh, b1, g1v, be1, b2, g2v, be2, b3, out);
}
